// Attention_68676527063657
// MI455X (gfx1250) — matrix operand order repaired and checked by reading, not compiled, not run on hardware
//
#include <hip/hip_runtime.h>
#include <hip/hip_bf16.h>

#define HEADS 16
#define DHEAD 64
#define DMODEL 1024
#define BATCH 2
#define SEQ 2048
#define BIGF 1e10f

typedef __attribute__((ext_vector_type(16))) __bf16 v16bf;
typedef __attribute__((ext_vector_type(8)))  __bf16 v8bf;
typedef __attribute__((ext_vector_type(8)))  float  v8f;
typedef __attribute__((ext_vector_type(4)))  int    v4i;

static __device__ __forceinline__ v16bf cat8(v8bf lo, v8bf hi) {
  return __builtin_shufflevector(lo, hi, 0,1,2,3,4,5,6,7,8,9,10,11,12,13,14,15);
}
static __device__ __forceinline__ v8f wmma_bf16(v16bf a, v16bf b, v8f c) {
  return __builtin_amdgcn_wmma_f32_16x16x32_bf16(false, a, false, b, (short)0, c,
                                                 false, false);
}

// ---- CDNA5 async global->LDS copy (ASYNCcnt), guarded for portability ------
#if __has_builtin(__builtin_amdgcn_global_load_async_to_lds_b128)
#define HAVE_ASYNC 1
typedef __attribute__((address_space(1))) v4i* as1_v4i_ptr;
typedef __attribute__((address_space(3))) v4i* as3_v4i_ptr;
static __device__ __forceinline__ void async_cp16(void* lds, const void* g) {
  __builtin_amdgcn_global_load_async_to_lds_b128(
      (as1_v4i_ptr)(g), (as3_v4i_ptr)(lds), 0, 0);
}
#else
#define HAVE_ASYNC 0
static __device__ __forceinline__ void async_cp16(void* lds, const void* g) {
  *(int4*)lds = *(const int4*)g;  // synchronous fallback
}
#endif

#if HAVE_ASYNC
#if __has_builtin(__builtin_amdgcn_s_wait_asynccnt)
#define WAIT_ASYNC(n) __builtin_amdgcn_s_wait_asynccnt(n)
#else
#define WAIT_ASYNC(n) asm volatile("s_wait_asynccnt %0" ::"i"(n) : "memory")
#endif
#else
#define WAIT_ASYNC(n) ((void)0)
#endif

// ---------------------------------------------------------------------------
// Projection GEMM: Y = X @ W^T  (X: [4096,1024] f32, W: [1024,1024] f32)
// Writes bf16:
//   z==0 -> Qw  [b,h,l,64]   z==1 -> Kw [b,h,l,64]   z==2 -> VwT [b,h,64,l]
// Workgroup tile 128(M) x 64(N), K-step 32 via LDS; 8 waves, wave = 16 rows.
// ---------------------------------------------------------------------------
__global__ __launch_bounds__(256) void proj_gemm_bf16_68676527(
    const float* __restrict__ q, const float* __restrict__ k,
    const float* __restrict__ v, const float* __restrict__ Wq,
    const float* __restrict__ Wk, const float* __restrict__ Wv,
    __bf16* __restrict__ Qw, __bf16* __restrict__ Kw, __bf16* __restrict__ VwT)
{
  __shared__ __align__(32) __bf16 As[128][32];  // X tile, row-major (m,k)
  __shared__ __align__(32) __bf16 Bs[64][32];   // Bs[n][kk] = W[n0+n][k0+kk]

  const int z = blockIdx.z;
  const float* X = (z == 0) ? q : (z == 1) ? k : v;
  const float* W = (z == 0) ? Wq : (z == 1) ? Wk : Wv;

  const int tid  = threadIdx.x;
  const int wave = tid >> 5;
  const int lane = tid & 31;
  const int ln   = lane & 15;
  const int hf   = lane >> 4;

  const int m0 = blockIdx.y * 128;  // row block in flattened [B*SEQ]
  const int n0 = blockIdx.x * 64;   // col block == head * 64

  const int arow = tid >> 1, ahalf = tid & 1;  // A: 128 rows x 2 col-halves
  const int brow = tid >> 2, bq    = tid & 3;  // B: 64 rows x 4 col-quarters

  v8f acc[4] = {};

  for (int k0 = 0; k0 < DMODEL; k0 += 32) {
    __syncthreads();
    {
      const float* src = X + (size_t)(m0 + arow) * DMODEL + k0 + ahalf * 16;
      v8bf t0, t1;
      #pragma unroll
      for (int i = 0; i < 8; i++) t0[i] = (__bf16)src[i];
      #pragma unroll
      for (int i = 0; i < 8; i++) t1[i] = (__bf16)src[8 + i];
      *(v8bf*)&As[arow][ahalf * 16]     = t0;
      *(v8bf*)&As[arow][ahalf * 16 + 8] = t1;
    }
    {
      const float* src = W + (size_t)(n0 + brow) * DMODEL + k0 + bq * 8;
      v8bf t0;
      #pragma unroll
      for (int i = 0; i < 8; i++) t0[i] = (__bf16)src[i];
      *(v8bf*)&Bs[brow][bq * 8] = t0;
    }
    __syncthreads();

    v16bf a = *(const v16bf*)&As[wave * 16 + ln][hf * 16];
    #pragma unroll
    for (int t = 0; t < 4; t++) {
      v16bf bfr = *(const v16bf*)&Bs[t * 16 + ln][hf * 16];
      acc[t] = wmma_bf16(a, bfr, acc[t]);
    }
  }

  // ---- epilogue: branch hoisted outside the store loops ----
  const int g_h  = blockIdx.x;                          // one head per N tile
  const int grow = m0 + wave * 16 + 8 * hf;             // + r below
  const int bb   = grow >> 11;
  const int ll0  = grow & (SEQ - 1);
  if (z == 2) {
    #pragma unroll
    for (int t = 0; t < 4; t++) {
      const int d = t * 16 + ln;
      __bf16* col = VwT + (((size_t)bb * HEADS + g_h) * DHEAD + d) * SEQ + ll0;
      #pragma unroll
      for (int r = 0; r < 8; r++) col[r] = (__bf16)acc[t][r];
    }
  } else {
    __bf16* dst = (z == 0) ? Qw : Kw;
    __bf16* base = dst + (((size_t)bb * HEADS + g_h) * SEQ + ll0) * DHEAD + ln;
    #pragma unroll
    for (int t = 0; t < 4; t++)
      #pragma unroll
      for (int r = 0; r < 8; r++)
        base[(size_t)r * DHEAD + t * 16] = (__bf16)acc[t][r];
  }
}

// ---------------------------------------------------------------------------
// Flash attention: 8 waves/WG, one 16-row q-tile per wave (128 q rows per WG).
// K/V tiles staged in LDS via async global->LDS copies, double-buffered.
// ---------------------------------------------------------------------------
__global__ __launch_bounds__(256) void flash_attn_bf16_68676527(
    const __bf16* __restrict__ Qw, const __bf16* __restrict__ Kw,
    const __bf16* __restrict__ VwT, const float* __restrict__ v_mask,
    const float* __restrict__ q_mask, float* __restrict__ out)
{
  __shared__ __align__(32) __bf16 Ks[2][32][64];  // k-tile rows x dk
  __shared__ __align__(32) __bf16 Vs[2][64][32];  // e x k-tile (from VwT)
  __shared__ __align__(32) __bf16 Pl[8][16][32];  // per-wave P staging

  const int tid  = threadIdx.x;
  const int wave = tid >> 5;
  const int lane = tid & 31;
  const int ln   = lane & 15;
  const int hf   = lane >> 4;

  const int b  = blockIdx.z;
  const int h  = blockIdx.y;
  const int bh = b * HEADS + h;
  const int q0 = blockIdx.x * 128 + wave * 16;

  // per-thread 16B slices of the cooperative K/V tile copies
  const int krow = tid >> 3, kcol = (tid & 7) * 8;  // Ks: 32 x 64
  const int vrow = tid >> 2, vcol = (tid & 3) * 8;  // Vs: 64 x 32
  const __bf16* Kbh = Kw + (size_t)bh * SEQ * DHEAD;
  const __bf16* Vbh = VwT + (size_t)bh * DHEAD * SEQ;

  // Q fragments for this wave (rows q0..q0+15, dk split into two K=32 frags)
  const __bf16* qrow = Qw + ((size_t)bh * SEQ + q0 + ln) * DHEAD;
  v16bf aq0 = *(const v16bf*)(qrow + hf * 16);
  v16bf aq1 = *(const v16bf*)(qrow + 32 + hf * 16);

  v8f o[4] = {};
  float mrow[8], lrow[8];
  #pragma unroll
  for (int r = 0; r < 8; r++) { mrow[r] = -1e30f; lrow[r] = 0.0f; }

  // prologue: issue tiles 0 and 1 (2 async ops per wave per tile)
  #pragma unroll
  for (int p = 0; p < 2; p++) {
    const int kp = p * 32;
    async_cp16(&Ks[p][krow][kcol], Kbh + (size_t)(kp + krow) * DHEAD + kcol);
    async_cp16(&Vs[p][vrow][vcol], Vbh + (size_t)vrow * SEQ + kp + vcol);
  }

  const int NT = SEQ / 32;  // 64
  for (int it = 0; it < NT; ++it) {
    const int buf = it & 1;
    const int k0  = it * 32;

    if (it + 1 < NT) WAIT_ASYNC(2);  // tile `it` landed; `it+1` may be in flight
    else             WAIT_ASYNC(0);
    __syncthreads();

    // ---- S = Q K^T : two 16x16 tiles, 4 WMMAs, B-frags from LDS ----
    v8f s[2];
    float pen[2];
    #pragma unroll
    for (int j = 0; j < 2; j++) {
      const int kk = k0 + 16 * j + ln;
      v16bf bk0 = *(const v16bf*)&Ks[buf][16 * j + ln][hf * 16];
      v16bf bk1 = *(const v16bf*)&Ks[buf][16 * j + ln][32 + hf * 16];
      v8f t = {};
      t = wmma_bf16(aq0, bk0, t);
      t = wmma_bf16(aq1, bk1, t);
      s[j]   = t;
      pen[j] = (1.0f - v_mask[b * SEQ + kk]) * BIGF;
    }

    // ---- scale + key padding mask + tril(incl diag) penalty ----
    #pragma unroll
    for (int j = 0; j < 2; j++) {
      const int kk = k0 + 16 * j + ln;
      #pragma unroll
      for (int r = 0; r < 8; r++) {
        const int qr = q0 + r + 8 * hf;
        float val = s[j][r] * 0.125f - pen[j];
        if (kk <= qr) val -= BIGF;  // reference quirk: diagonal masked too
        s[j][r] = val;
      }
    }

    // ---- online softmax (row stats replicated across 16 lanes per half) ----
    float alpha[8];
    #pragma unroll
    for (int r = 0; r < 8; r++) {
      float t = fmaxf(s[0][r], s[1][r]);
      #pragma unroll
      for (int off = 1; off < 16; off <<= 1) t = fmaxf(t, __shfl_xor(t, off, 32));
      const float mnew = fmaxf(mrow[r], t);
      alpha[r] = __expf(mrow[r] - mnew);
      mrow[r]  = mnew;
      const float p0 = __expf(s[0][r] - mnew);
      const float p1 = __expf(s[1][r] - mnew);
      s[0][r] = p0;
      s[1][r] = p1;
      float ssum = p0 + p1;
      #pragma unroll
      for (int off = 1; off < 16; off <<= 1) ssum += __shfl_xor(ssum, off, 32);
      lrow[r] = lrow[r] * alpha[r] + ssum;
    }
    #pragma unroll
    for (int t4 = 0; t4 < 4; t4++)
      #pragma unroll
      for (int r = 0; r < 8; r++) o[t4][r] *= alpha[r];

    // ---- C-layout -> A-layout for P via wave-private LDS ----
    #pragma unroll
    for (int j = 0; j < 2; j++)
      #pragma unroll
      for (int r = 0; r < 8; r++)
        Pl[wave][r + 8 * hf][16 * j + ln] = (__bf16)s[j][r];
    v16bf ap = *(const v16bf*)&Pl[wave][ln][hf * 16];

    // ---- O += P V : 4 WMMAs, B-frags from LDS V tile ----
    #pragma unroll
    for (int t4 = 0; t4 < 4; t4++) {
      v16bf bv = *(const v16bf*)&Vs[buf][16 * t4 + ln][hf * 16];
      o[t4] = wmma_bf16(ap, bv, o[t4]);
    }

    __syncthreads();  // everyone done reading buf before it is overwritten
    if (it + 2 < NT) {
      const int kp = (it + 2) * 32;
      async_cp16(&Ks[buf][krow][kcol], Kbh + (size_t)(kp + krow) * DHEAD + kcol);
      async_cp16(&Vs[buf][vrow][vcol], Vbh + (size_t)vrow * SEQ + kp + vcol);
    }
  }

  // ---- epilogue: normalize, apply q_mask, store f32 ----
  float inv[8];
  #pragma unroll
  for (int r = 0; r < 8; r++) {
    const int qr = q0 + r + 8 * hf;
    inv[r] = q_mask[b * SEQ + qr] / lrow[r];
  }
  #pragma unroll
  for (int t4 = 0; t4 < 4; t4++)
    #pragma unroll
    for (int r = 0; r < 8; r++) {
      const int qr = q0 + r + 8 * hf;
      out[((size_t)b * SEQ + qr) * DMODEL + h * DHEAD + t4 * 16 + ln] =
          o[t4][r] * inv[r];
    }
}

extern "C" void kernel_launch(void* const* d_in, const int* in_sizes, int n_in,
                              void* d_out, int out_size, void* d_ws,
                              size_t ws_size, hipStream_t stream) {
  (void)in_sizes; (void)n_in; (void)out_size; (void)ws_size;
  const float* q      = (const float*)d_in[0];
  const float* k      = (const float*)d_in[1];
  const float* v      = (const float*)d_in[2];
  const float* v_mask = (const float*)d_in[3];
  const float* q_mask = (const float*)d_in[4];
  const float* Wq     = (const float*)d_in[5];
  const float* Wk     = (const float*)d_in[6];
  const float* Wv     = (const float*)d_in[7];

  const size_t per = (size_t)BATCH * HEADS * SEQ * DHEAD;  // 4M bf16 elems
  __bf16* Qw  = (__bf16*)d_ws;
  __bf16* Kw  = Qw + per;
  __bf16* VwT = Kw + per;  // stored [b,h,64,SEQ]  (needs 24 MB of d_ws)

  dim3 pgrid(DMODEL / 64, (BATCH * SEQ) / 128, 3);
  proj_gemm_bf16_68676527<<<pgrid, 256, 0, stream>>>(q, k, v, Wq, Wk, Wv,
                                                     Qw, Kw, VwT);

  dim3 agrid(SEQ / 128, HEADS, BATCH);
  flash_attn_bf16_68676527<<<agrid, 256, 0, stream>>>(Qw, Kw, VwT, v_mask,
                                                      q_mask, (float*)d_out);
}
